// StarAttention_3994319586083
// MI455X (gfx1250) — compile-verified, matrix operand order checked by reading, not run on hardware
//
#include <hip/hip_runtime.h>

typedef _Float16 half8  __attribute__((ext_vector_type(8)));
typedef _Float16 half16 __attribute__((ext_vector_type(16)));
typedef float    f32x8  __attribute__((ext_vector_type(8)));

#define HIDDEN 1024
#define HEADS  16
#define HD     64
#define BLK    2048
#define BATCH  4
#define SQ     2048
#define MTOK   8192   // BATCH * 2048 tokens

// 0.125 (=1/sqrt(64)) * log2(e): base-2 softmax domain (v_exp_f32 is 2^x)
#define SCALE_LOG2E 0.1803368801111204f

// ---------------------------------------------------------------------------
// Fragment load for 16-bit WMMA A/B operands from a row-major f16 matrix.
// CDNA5 16-bit 16x32 layout: lane holds row (lane&15); lanes 0-15 carry
// K = {k0..k0+7, k0+16..k0+23}, lanes 16-31 carry K = {k0+8..k0+15, k0+24..31}.
// => two contiguous 16-byte chunks per lane at k0+8*g and k0+16+8*g.
// ---------------------------------------------------------------------------
__device__ __forceinline__ half16 load_frag(const _Float16* base, int ld,
                                            int row0, int k0, int lane) {
  const int n = lane & 15, g = lane >> 4;
  const _Float16* p = base + (size_t)(row0 + n) * ld + k0 + 8 * g;
  union { half16 v; half8 h[2]; } u;
  u.h[0] = *(const half8*)(p);
  u.h[1] = *(const half8*)(p + 16);
  return u.v;
}

__device__ __forceinline__ f32x8 wmma_f16(half16 a, half16 b, f32x8 c) {
  return __builtin_amdgcn_wmma_f32_16x16x32_f16(false, a, false, b,
                                                (short)0, c, false, false);
}

__device__ __forceinline__ float rmax16(float x) {
#pragma unroll
  for (int m = 8; m >= 1; m >>= 1) x = fmaxf(x, __shfl_xor(x, m, 32));
  return x;
}
__device__ __forceinline__ float rsum16(float x) {
#pragma unroll
  for (int m = 8; m >= 1; m >>= 1) x += __shfl_xor(x, m, 32);
  return x;
}

// ---------------------------------------------------------------------------
// LayerNorm(f32 in) -> f16 out. One token per 256-thread block (4 f32/lane).
// slice=1 selects input_ids[:, host_id*BLK : host_id*BLK+BLK].
// ---------------------------------------------------------------------------
__global__ __launch_bounds__(256) void ln_kernel(
    const float* __restrict__ x, const float* __restrict__ gamma,
    const float* __restrict__ beta, const int* __restrict__ host_id,
    int slice, _Float16* __restrict__ out)
{
  const int token = blockIdx.x;
  size_t src_row;
  if (slice) {
    const int bb = token >> 11, i = token & 2047;
    src_row = (size_t)bb * 8192 + (size_t)host_id[0] * BLK + i;
  } else {
    src_row = (size_t)token;
  }
  const float4 v = ((const float4*)(x + src_row * HIDDEN))[threadIdx.x];
  float s  = v.x + v.y + v.z + v.w;
  float ss = v.x * v.x + v.y * v.y + v.z * v.z + v.w * v.w;
#pragma unroll
  for (int m = 16; m >= 1; m >>= 1) {
    s  += __shfl_xor(s,  m, 32);
    ss += __shfl_xor(ss, m, 32);
  }
  __shared__ float red[2][8];
  const int wave = threadIdx.x >> 5, lane = threadIdx.x & 31;
  if (lane == 0) { red[0][wave] = s; red[1][wave] = ss; }
  __syncthreads();
  s = 0.f; ss = 0.f;
#pragma unroll
  for (int w = 0; w < 8; ++w) { s += red[0][w]; ss += red[1][w]; }
  const float mu  = s * (1.0f / HIDDEN);
  const float var = ss * (1.0f / HIDDEN) - mu * mu;
  const float rs  = rsqrtf(var + 1e-12f);
  const float4 gm = ((const float4*)gamma)[threadIdx.x];
  const float4 bt = ((const float4*)beta)[threadIdx.x];
  _Float16* orow = out + (size_t)token * HIDDEN;
  const int base = threadIdx.x * 4;
  orow[base + 0] = (_Float16)((v.x - mu) * rs * gm.x + bt.x);
  orow[base + 1] = (_Float16)((v.y - mu) * rs * gm.y + bt.y);
  orow[base + 2] = (_Float16)((v.z - mu) * rs * gm.z + bt.z);
  orow[base + 3] = (_Float16)((v.w - mu) * rs * gm.w + bt.w);
}

// ---------------------------------------------------------------------------
// W[k][n] f32  ->  WT[n][k] f16   (LDS-tiled 32x32 transpose, coalesced)
// ---------------------------------------------------------------------------
__global__ __launch_bounds__(256) void transpose_w(const float* __restrict__ W,
                                                   _Float16* __restrict__ WT)
{
  __shared__ float tile[32][33];
  const int bx = blockIdx.x * 32;   // k base
  const int by = blockIdx.y * 32;   // n base
  const int tx = threadIdx.x & 31, ty = threadIdx.x >> 5;
#pragma unroll
  for (int i = 0; i < 32; i += 8)
    tile[ty + i][tx] = W[(size_t)(bx + ty + i) * HIDDEN + by + tx];
  __syncthreads();
#pragma unroll
  for (int i = 0; i < 32; i += 8)
    WT[(size_t)(by + ty + i) * HIDDEN + bx + tx] = (_Float16)tile[tx][ty + i];
}

// ---------------------------------------------------------------------------
// GEMM: out[M x 1024] = A[M x 1024] * WT^T + bias, f16 in / f32 acc / f16 out.
// Block = 8 waves -> 64x256 tile; each wave owns 32x64 (8 WMMA accumulators).
// mode 0: row-major out[row][col].
// mode 1: V-transposed out: VT[((b*16+h)*64+d)][t], b=row/2048, t=row%2048.
// ---------------------------------------------------------------------------
__global__ __launch_bounds__(256) void gemm_kernel(
    const _Float16* __restrict__ A, const _Float16* __restrict__ WT,
    const float* __restrict__ bias, _Float16* __restrict__ out, int mode)
{
  const int wave = threadIdx.x >> 5, lane = threadIdx.x & 31;
  const int n = lane & 15, g = lane >> 4;
  const int wm = blockIdx.x * 64  + (wave & 1) * 32;
  const int wn = blockIdx.y * 256 + (wave >> 1) * 64;

  f32x8 acc[2][4] = {};
  for (int k = 0; k < HIDDEN; k += 32) {
    half16 a0 = load_frag(A, HIDDEN, wm,      k, lane);
    half16 a1 = load_frag(A, HIDDEN, wm + 16, k, lane);
#pragma unroll
    for (int j = 0; j < 4; ++j) {
      half16 bfr = load_frag(WT, HIDDEN, wn + 16 * j, k, lane);
      acc[0][j] = wmma_f16(a0, bfr, acc[0][j]);
      acc[1][j] = wmma_f16(a1, bfr, acc[1][j]);
    }
  }
#pragma unroll
  for (int i = 0; i < 2; ++i)
#pragma unroll
    for (int j = 0; j < 4; ++j) {
      const int col = wn + 16 * j + n;
      const float bv = bias[col];
#pragma unroll
      for (int r = 0; r < 8; ++r) {
        const int row = wm + 16 * i + r + 8 * g;   // D tile: m = r + 8*g
        const float val = acc[i][j][r] + bv;
        if (mode == 0) {
          out[(size_t)row * HIDDEN + col] = (_Float16)val;
        } else {
          const int bb = row >> 11, t = row & 2047;
          const int hh = col >> 6,  d = col & 63;
          out[((size_t)((bb * HEADS + hh) * HD + d)) * BLK + t] = (_Float16)val;
        }
      }
    }
}

// ---------------------------------------------------------------------------
// Flash attention: block = (b, h, 128 q-rows), 8 waves x 16 q-rows each.
// 64-key steps: 8 score WMMAs + base-2 online softmax + LDS D->A transpose +
// 8 context WMMAs per step (16 WMMAs / step, softmax bookkeeping amortized).
// ---------------------------------------------------------------------------
__global__ __launch_bounds__(256) void attn_kernel(
    const _Float16* __restrict__ Q16, const _Float16* __restrict__ K16,
    const _Float16* __restrict__ VT16, float* __restrict__ out)
{
  __shared__ __align__(16) _Float16 plds[8 * 16 * 64];   // 2KB per wave
  const int wave = threadIdx.x >> 5, lane = threadIdx.x & 31;
  const int n = lane & 15, g = lane >> 4;
  const int b = blockIdx.z, h = blockIdx.y;
  const int qbase = blockIdx.x * 128 + wave * 16;
  const int qrow0 = b * SQ + qbase;
  const int hoff  = h * HD;
  const int vrow0 = (b * HEADS + h) * HD;

  const half16 qa0 = load_frag(Q16, HIDDEN, qrow0, hoff + 0,  lane);
  const half16 qa1 = load_frag(Q16, HIDDEN, qrow0, hoff + 32, lane);

  f32x8 cacc[4] = {};
  float m_run[8], s_run[8];
#pragma unroll
  for (int r = 0; r < 8; ++r) { m_run[r] = -3.0e38f; s_run[r] = 0.f; }

  _Float16* pl = plds + wave * (16 * 64);

  for (int kb = 0; kb < BLK; kb += 64) {
    const int krow = b * BLK + kb;

    // ---- scores: 4 tiles of 16 keys, S = (Q.K^T) * (0.125*log2e) ----
    f32x8 S[4];
#pragma unroll
    for (int j = 0; j < 4; ++j) {
      half16 ka = load_frag(K16, HIDDEN, krow + 16 * j, hoff + 0,  lane);
      half16 kbf = load_frag(K16, HIDDEN, krow + 16 * j, hoff + 32, lane);
      f32x8 sj = {};
      sj = wmma_f16(qa0, ka, sj);
      sj = wmma_f16(qa1, kbf, sj);
      S[j] = sj;
    }

    // ---- base-2 online softmax (row stats replicated per 16-lane group) ----
    float alpha[8];
#pragma unroll
    for (int r = 0; r < 8; ++r) {
      float s0 = S[0][r] * SCALE_LOG2E;
      float s1 = S[1][r] * SCALE_LOG2E;
      float s2 = S[2][r] * SCALE_LOG2E;
      float s3 = S[3][r] * SCALE_LOG2E;
      const float tm = rmax16(fmaxf(fmaxf(s0, s1), fmaxf(s2, s3)));
      const float mn = fmaxf(m_run[r], tm);
      const float a  = __builtin_amdgcn_exp2f(m_run[r] - mn);
      const float p0 = __builtin_amdgcn_exp2f(s0 - mn);
      const float p1 = __builtin_amdgcn_exp2f(s1 - mn);
      const float p2 = __builtin_amdgcn_exp2f(s2 - mn);
      const float p3 = __builtin_amdgcn_exp2f(s3 - mn);
      const float ts = rsum16((p0 + p1) + (p2 + p3));
      s_run[r] = s_run[r] * a + ts;
      m_run[r] = mn;
      alpha[r] = a;
      S[0][r] = p0; S[1][r] = p1; S[2][r] = p2; S[3][r] = p3;
    }
#pragma unroll
    for (int t = 0; t < 4; ++t)
#pragma unroll
      for (int r = 0; r < 8; ++r) cacc[t][r] *= alpha[r];

    // ---- D-layout (lane=col) -> A-fragment (lane=row) via LDS ----
#pragma unroll
    for (int r = 0; r < 8; ++r) {
      const int m = r + 8 * g;
#pragma unroll
      for (int j = 0; j < 4; ++j)
        pl[m * 64 + 16 * j + n] = (_Float16)S[j][r];
    }
    __syncthreads();
    const half16 pa0 = load_frag(pl, 64, 0, 0,  lane);
    const half16 pa1 = load_frag(pl, 64, 0, 32, lane);

    // ---- ctx += P . V  (V^T gives contiguous key-dim chunks) ----
#pragma unroll
    for (int t = 0; t < 4; ++t) {
      half16 vb0 = load_frag(VT16, BLK, vrow0 + 16 * t, kb,      lane);
      half16 vb1 = load_frag(VT16, BLK, vrow0 + 16 * t, kb + 32, lane);
      cacc[t] = wmma_f16(pa0, vb0, cacc[t]);
      cacc[t] = wmma_f16(pa1, vb1, cacc[t]);
    }
    __syncthreads();
  }

  // ctx / row_sum  ->  out[b, q, h*64 + d]  (f32)
#pragma unroll
  for (int t = 0; t < 4; ++t) {
#pragma unroll
    for (int r = 0; r < 8; ++r) {
      const int m = r + 8 * g;
      const int q = qbase + m;
      const int d = hoff + 16 * t + n;
      out[((size_t)(b * SQ + q)) * HIDDEN + d] = cacc[t][r] / s_run[r];
    }
  }
}

// ---------------------------------------------------------------------------
extern "C" void kernel_launch(void* const* d_in, const int* in_sizes, int n_in,
                              void* d_out, int out_size, void* d_ws, size_t ws_size,
                              hipStream_t stream) {
  const float* input_ids = (const float*)d_in[0];
  const float* query     = (const float*)d_in[1];
  const float* Wq = (const float*)d_in[2];
  const float* bq = (const float*)d_in[3];
  const float* Wk = (const float*)d_in[4];
  const float* bk = (const float*)d_in[5];
  const float* Wv = (const float*)d_in[6];
  const float* bv = (const float*)d_in[7];
  const float* ln_g = (const float*)d_in[8];
  const float* ln_b = (const float*)d_in[9];
  const int* host_id = (const int*)d_in[10];
  float* out = (float*)d_out;

  char* ws = (char*)d_ws;
  size_t off = 0;
  auto alloc = [&](size_t bytes) {
    char* p = ws + off;
    off += (bytes + 255) & ~(size_t)255;
    return p;
  };
  _Float16* qh   = (_Float16*)alloc((size_t)MTOK * HIDDEN * 2);   // 16 MB
  _Float16* lh   = (_Float16*)alloc((size_t)MTOK * HIDDEN * 2);   // 16 MB
  _Float16* WqT  = (_Float16*)alloc((size_t)HIDDEN * HIDDEN * 2); //  2 MB
  _Float16* WkT  = (_Float16*)alloc((size_t)HIDDEN * HIDDEN * 2);
  _Float16* WvT  = (_Float16*)alloc((size_t)HIDDEN * HIDDEN * 2);
  _Float16* Q16  = (_Float16*)alloc((size_t)MTOK * HIDDEN * 2);   // 16 MB
  _Float16* K16  = (_Float16*)alloc((size_t)MTOK * HIDDEN * 2);   // 16 MB
  _Float16* VT16 = (_Float16*)alloc((size_t)MTOK * HIDDEN * 2);   // 16 MB

  // 1) LayerNorm -> f16 activations
  ln_kernel<<<MTOK, 256, 0, stream>>>(query,     ln_g, ln_b, host_id, 0, qh);
  ln_kernel<<<MTOK, 256, 0, stream>>>(input_ids, ln_g, ln_b, host_id, 1, lh);

  // 2) Weight transposes -> f16 [N][K]
  transpose_w<<<dim3(32, 32), 256, 0, stream>>>(Wq, WqT);
  transpose_w<<<dim3(32, 32), 256, 0, stream>>>(Wk, WkT);
  transpose_w<<<dim3(32, 32), 256, 0, stream>>>(Wv, WvT);

  // 3) QKV projections (V stored transposed per (b,h) for the P.V WMMA)
  gemm_kernel<<<dim3(128, 4), 256, 0, stream>>>(qh, WqT, bq, Q16, 0);
  gemm_kernel<<<dim3(128, 4), 256, 0, stream>>>(lh, WkT, bk, K16, 0);
  gemm_kernel<<<dim3(128, 4), 256, 0, stream>>>(lh, WvT, bv, VT16, 1);

  // 4) Flash attention
  attn_kernel<<<dim3(SQ / 128, HEADS, BATCH), 256, 0, stream>>>(Q16, K16, VT16, out);
}
